// GeneralMPGNN_18837726561017
// MI455X (gfx1250) — compile-verified, matrix operand order checked by reading, not run on hardware
//
#include <hip/hip_runtime.h>
#include <math.h>

// ---------------------------------------------------------------------------
// GeneralMPGNN on MI455X (gfx1250). All MLPs via v_wmma_f32_16x16x32_f16
// (f16 in, f32 accumulate); one wave32 owns a 16-row tile. Segment mean/max
// via f32 atomicAdd + integer-bit atomicMax (post-ReLU values >= 0).
//
// LDS operands live in WMMA fragment order: each A/B fragment is one
// contiguous 32-byte region (2x ds_load_b128). Staging is branch-free:
// compile-time 8-column spans, each lane stores the 4-group at
// c0 = 8t + 4*half as one ds_store_b64; fragment address is affine in half:
//   idx = ((t>>2)*32 + r + 16*(t&1))*16 + (t&2)*4 + 4*half.
// Hidden layer columns are permuted (n -> 2n, n+16 -> 2n+1; W2 rows permuted
// to match) so each lane's two D values pack into one ds_store_b32.
//   C/D (16x16 f32): VGPR j -> row j + 8*(lane>>4), col = lane&15.
// ---------------------------------------------------------------------------

typedef __attribute__((ext_vector_type(16))) _Float16 v16h;
typedef __attribute__((ext_vector_type(4)))  _Float16 h4v;
typedef __attribute__((ext_vector_type(2)))  _Float16 h2v;
typedef __attribute__((ext_vector_type(8)))  float    v8f;

#define TB  256
#define WPB 8   // waves per block (wave32)

#define WMMA_F16(a, b, c) \
  __builtin_amdgcn_wmma_f32_16x16x32_f16(false, (a), false, (b), (short)0, (c), false, false)

// (row r, column c) -> fragment-order index (generic form).
__device__ __forceinline__ int frag_idx(int r, int c) {
  int kc = c >> 5, k = c & 31, h = (k >> 3) & 1;
  int j = (k < 16) ? (k - 8 * h) : (8 + (k - 16) - 8 * h);
  return (kc * 32 + r + 16 * h) * 16 + j;
}

// Fragment index of the 4-group at c0 = 8t + 4*half (closed form, no branches).
__device__ __forceinline__ int span_idx(int t, int r, int half) {
  return ((t >> 2) * 32 + r + 16 * (t & 1)) * 16 + ((t & 2) << 2) + 4 * half;
}

__device__ __forceinline__ void stage_one(_Float16* f, int r, int c, float v) {
  f[frag_idx(r, c)] = (_Float16)v;
}

__device__ __forceinline__ void sp_store(_Float16* at, int t, int r, int half,
                                         float a, float b, float c, float d) {
  h4v q;
  q[0] = (_Float16)a; q[1] = (_Float16)b; q[2] = (_Float16)c; q[3] = (_Float16)d;
  *(h4v*)&at[span_idx(t, r, half)] = q;
}

__device__ __forceinline__ void sp4f(_Float16* at, int t, int r, int half, const float* p) {
  sp_store(at, t, r, half, p[0], p[1], p[2], p[3]);
}

__device__ __forceinline__ void sp4s(_Float16* at, int t, int r, int half,
                                     const float* p, float s) {
  sp_store(at, t, r, half, p[0] * s, p[1] * s, p[2] * s, p[3] * s);
}

__device__ __forceinline__ void sp0(_Float16* at, int t, int r, int half) {
  h4v q = {};
  *(h4v*)&at[span_idx(t, r, half)] = q;
}

template <typename F>
__device__ __forceinline__ void spf(_Float16* at, int t, int r, int half, int c0, F&& src) {
  sp_store(at, t, r, half, src(c0), src(c0 + 1), src(c0 + 2), src(c0 + 3));
}

// W1 [fin x 32] f32 -> frag layout [2 ntiles][KC kchunks][32 lanes][16] f16.
__device__ __forceinline__ void stage_w1_frag(const float* W1, int fin, int KC,
                                              _Float16* w1F) {
  int total = 2 * KC * 512;
  for (int i = threadIdx.x; i < total; i += TB) {
    int j = i & 15;
    int lane = (i >> 4) & 31;
    int kc = (i >> 9) % KC;
    int nt = i / (KC * 512);
    int h = lane >> 4;
    int n = (lane & 15) + 16 * nt;
    int kw = (j < 8) ? (j + 8 * h) : (16 + (j - 8) + 8 * h);
    int k = kc * 32 + kw;
    w1F[i] = (_Float16)((k < fin) ? W1[k * 32 + n] : 0.f);
  }
}

// W2 [32 x fout] f32 -> frag layout [32 lanes][16] f16 (cols >= fout zero),
// with hidden-column permutation: fragment k-slot k holds original W2 row
// ko = (k>>1) + 16*(k&1)  (inverse of n->2n, n+16->2n+1).
__device__ __forceinline__ void stage_w2_frag(const float* W2, const float* B1,
                                              const float* B2, int fout,
                                              _Float16* w2F, float* b1s, float* b2s) {
  for (int i = threadIdx.x; i < 512; i += TB) {
    int j = i & 15, lane = i >> 4;
    int h = lane >> 4, n = lane & 15;
    int k = (j < 8) ? (j + 8 * h) : (16 + (j - 8) + 8 * h);
    int ko = (k >> 1) + ((k & 1) << 4);
    w2F[i] = (_Float16)((n < fout) ? W2[ko * fout + n] : 0.f);
  }
  if (threadIdx.x < 32) b1s[threadIdx.x] = B1[threadIdx.x];
  if (threadIdx.x < 16) b2s[threadIdx.x] = (threadIdx.x < (unsigned)fout) ? B2[threadIdx.x] : 0.f;
}

// 2-layer MLP on a 16-row tile staged in fragment order. KC = K/32 chunks of
// the first GEMM. Returns D (lane: col lane&15, rows v + 8*(lane>>4)).
template <int KC, bool RELU2>
__device__ __forceinline__ v8f mlp_tile(const _Float16* atF, _Float16* htF,
                                        const _Float16* w1F, const float* b1,
                                        const _Float16* w2F, const float* b2,
                                        int lane) {
  const int half = lane >> 4, n = lane & 15;
  const v16h* av  = (const v16h*)atF;
  const v16h* w1v = (const v16h*)w1F;
  const v16h* w2v = (const v16h*)w2F;
  v8f c0 = {}, c1 = {};
#pragma unroll
  for (int kc = 0; kc < KC; ++kc) {
    v16h a = av[kc * 32 + lane];
    c0 = WMMA_F16(a, w1v[(0 * KC + kc) * 32 + lane], c0);
    c1 = WMMA_F16(a, w1v[(1 * KC + kc) * 32 + lane], c1);
  }
  // bias + relu; write hidden in A-fragment order under the column
  // permutation n->2n, n+16->2n+1: both values land in adjacent slots of one
  // target lane -> single packed ds_store_b32 per row.
  const int k2 = 2 * n;
  const int hh = (k2 >> 3) & 1;
  const int j2 = (k2 < 16) ? (k2 - 8 * hh) : (8 + (k2 - 16) - 8 * hh);
#pragma unroll
  for (int v = 0; v < 8; ++v) {
    int m = v + 8 * half;
    float h0 = fmaxf(c0[v] + b1[n], 0.f);
    float h1 = fmaxf(c1[v] + b1[n + 16], 0.f);
    h2v p;
    p[0] = (_Float16)h0;
    p[1] = (_Float16)h1;
    *(h2v*)&htF[(m + 16 * hh) * 16 + j2] = p;
  }
  asm volatile("s_wait_dscnt 0" ::: "memory");  // cross-lane LDS reshuffle
  v8f d = {};
  {
    v16h a2 = ((const v16h*)htF)[lane];
    d = WMMA_F16(a2, w2v[lane], d);
  }
#pragma unroll
  for (int v = 0; v < 8; ++v) {
    float t2 = d[v] + b2[n];
    if (RELU2) t2 = fmaxf(t2, 0.f);
    d[v] = t2;
  }
  return d;
}

__global__ void build_u4_kernel(const float* __restrict__ u,
                                const float* __restrict__ prm,
                                float* __restrict__ u4, int G) {
  int g = blockIdx.x * blockDim.x + threadIdx.x;
  if (g < G) {
    u4[g * 4 + 0] = u[g * 3 + 0];
    u4[g * 4 + 1] = u[g * 3 + 1];
    u4[g * 4 + 2] = u[g * 3 + 2];
    u4[g * 4 + 3] = prm[g];
  }
}

// Fused EdgeModel + NodeModel-message kernel.
template <int XF, int EAF, int UF, int K1>
__global__ __launch_bounds__(TB) void edge_layer_kernel(
    const float* __restrict__ x, const int* __restrict__ ei,
    const float* __restrict__ ea_in, const float* __restrict__ uu,
    const int* __restrict__ batch,
    const float* __restrict__ eW1, const float* __restrict__ eB1,
    const float* __restrict__ eW2, const float* __restrict__ eB2,
    const float* __restrict__ mW1, const float* __restrict__ mB1,
    const float* __restrict__ mW2, const float* __restrict__ mB2,
    float* __restrict__ ea_out, float* __restrict__ nsum,
    float* __restrict__ nmax, float* __restrict__ ncnt, int E) {
  constexpr int KCE = K1 / 32;
  __shared__ __align__(32) _Float16 sW1e[2 * KCE * 512], sW2e[512];
  __shared__ __align__(32) _Float16 sW1m[2 * 512], sW2m[512];
  __shared__ float sB1e[32], sB2e[16], sB1m[32], sB2m[16];
  __shared__ __align__(32) _Float16 sT[WPB][KCE * 512];
  __shared__ __align__(32) _Float16 sH[WPB][512];

  stage_w1_frag(eW1, 2 * XF + EAF + UF, KCE, sW1e);
  stage_w2_frag(eW2, eB1, eB2, 6, sW2e, sB1e, sB2e);
  stage_w1_frag(mW1, 2 * XF + 6, 1, sW1m);
  stage_w2_frag(mW2, mB1, mB2, 12, sW2m, sB1m, sB2m);
  __syncthreads();

  const int lane = threadIdx.x & 31, wv = threadIdx.x >> 5;
  const int r = lane & 15, half = lane >> 4, nc = lane & 15;
  const int ntiles = (E + 15) >> 4;
  _Float16* at = sT[wv];

  for (int tile = blockIdx.x * WPB + wv; tile < ntiles; tile += gridDim.x * WPB) {
    const int base = tile << 4;
    const int e = base + r;
    const bool ve = (e < E);
    int row = 0, col = 0, g = 0, eS = 0;
    if (ve) { row = ei[e]; col = ei[E + e]; g = batch[row]; eS = e; }

    constexpr int finE = 2 * XF + EAF + UF;
    auto ein = [&](int c) -> float {     // straddle fallback (clamped indices)
      if (c >= finE) return 0.f;
      if (c < XF)            return x[(size_t)row * XF + c];
      if (c < 2 * XF)        return x[(size_t)col * XF + (c - XF)];
      if (c < 2 * XF + EAF)  return ea_in[(size_t)eS * EAF + (c - 2 * XF)];
      return uu[g * UF + (c - 2 * XF - EAF)];
    };
    // stage edge-MLP input [x[row] | x[col] | ea | u[g]] (zero-padded to K1)
#pragma unroll
    for (int t = 0; t < K1 / 8; ++t) {
      const int lo = 8 * t, hi = lo + 8;
      const int c0 = lo + 4 * half;
      if (hi <= XF)                               sp4f(at, t, r, half, x + (size_t)row * XF + c0);
      else if (lo >= XF && hi <= 2 * XF)          sp4f(at, t, r, half, x + (size_t)col * XF + (c0 - XF));
      else if (lo >= 2 * XF && hi <= 2 * XF + EAF) sp4f(at, t, r, half, ea_in + (size_t)eS * EAF + (c0 - 2 * XF));
      else if (lo >= finE)                        sp0(at, t, r, half);
      else                                        spf(at, t, r, half, c0, ein);
    }
    asm volatile("s_wait_dscnt 0" ::: "memory");
    v8f de = mlp_tile<KCE, true>(at, sH[wv], sW1e, sB1e, sW2e, sB2e, lane);

    // store ea'; stage ea' (+2 pad cols) into the message tile (frag, KC=1)
    if (nc < 8) {
#pragma unroll
      for (int v = 0; v < 8; ++v) {
        int m = v + 8 * half;
        int ee = base + m;
        float val = (nc < 6) ? de[v] : 0.f;
        stage_one(at, m, 2 * XF + nc, val);
        if (nc < 6 && ee < E) ea_out[(size_t)ee * 6 + nc] = val;
      }
    }
    // restage x spans for the message MLP ([2XF, 2XF+8) written above)
    auto xin = [&](int c) -> float {
      return (c < XF) ? x[(size_t)row * XF + c] : x[(size_t)col * XF + (c - XF)];
    };
#pragma unroll
    for (int t = 0; t < 4; ++t) {
      const int lo = 8 * t, hi = lo + 8;
      const int c0 = lo + 4 * half;
      if (hi <= XF)                       sp4f(at, t, r, half, x + (size_t)row * XF + c0);
      else if (lo >= XF && hi <= 2 * XF)  sp4f(at, t, r, half, x + (size_t)col * XF + (c0 - XF));
      else if (hi <= 2 * XF)              spf(at, t, r, half, c0, xin);   // x row/col straddle
      else if (lo >= 2 * XF + 8)          sp0(at, t, r, half);
      // span [2XF, 2XF+8): ea' + pads, already staged by the D path
    }
    asm volatile("s_wait_dscnt 0" ::: "memory");
    v8f dm = mlp_tile<1, true>(at, sH[wv], sW1m, sB1m, sW2m, sB2m, lane);

    if (nc < 12) {
#pragma unroll
      for (int v = 0; v < 8; ++v) {
        int m = v + 8 * half;
        int ee = base + m;
        if (ee < E) {
          int cc = ei[E + ee];
          float val = dm[v];
          atomicAdd(&nsum[(size_t)cc * 12 + nc], val);
          atomicMax((int*)&nmax[(size_t)cc * 12 + nc], __float_as_int(val));
        }
      }
    }
    if (half == 0 && ve) atomicAdd(&ncnt[col], 1.0f);
  }
}

// NodeModel second MLP + per-graph pooling scatter.
template <int XF, int UF, int KN>
__global__ __launch_bounds__(TB) void node_layer_kernel(
    const float* __restrict__ x, const float* __restrict__ nsum,
    const float* __restrict__ nmax, const float* __restrict__ ncnt,
    const float* __restrict__ uu, const int* __restrict__ batch,
    const float* __restrict__ W1, const float* __restrict__ B1,
    const float* __restrict__ W2, const float* __restrict__ B2,
    float* __restrict__ x_out, float* __restrict__ gsum,
    float* __restrict__ gmax, float* __restrict__ gcnt, int N) {
  constexpr int KC = KN / 32;
  __shared__ __align__(32) _Float16 sW1[2 * KC * 512], sW2[512];
  __shared__ float sB1[32], sB2[16];
  __shared__ __align__(32) _Float16 sT[WPB][KC * 512];
  __shared__ __align__(32) _Float16 sH[WPB][512];
  stage_w1_frag(W1, XF + 24 + UF, KC, sW1);
  stage_w2_frag(W2, B1, B2, 12, sW2, sB1, sB2);
  __syncthreads();

  const int lane = threadIdx.x & 31, wv = threadIdx.x >> 5;
  const int r = lane & 15, half = lane >> 4, nc = lane & 15;
  const int ntiles = (N + 15) >> 4;
  _Float16* at = sT[wv];

  for (int tile = blockIdx.x * WPB + wv; tile < ntiles; tile += gridDim.x * WPB) {
    const int base = tile << 4;
    const int nid = base + r;
    const bool vn = (nid < N);
    const int nidS = vn ? nid : 0;
    int g = vn ? batch[nidS] : 0;
    float rcnt = 1.0f / fmaxf(ncnt[nidS], 1.f);
    constexpr int fin = XF + 24 + UF;
    auto nin = [&](int c) -> float {
      if (c >= fin) return 0.f;
      if (c < XF)       return x[(size_t)nidS * XF + c];
      if (c < XF + 12)  return nsum[(size_t)nidS * 12 + (c - XF)] * rcnt;
      if (c < XF + 24)  return nmax[(size_t)nidS * 12 + (c - XF - 12)];
      return uu[g * UF + (c - XF - 24)];
    };
#pragma unroll
    for (int t = 0; t < KN / 8; ++t) {
      const int lo = 8 * t, hi = lo + 8;
      const int c0 = lo + 4 * half;
      if (hi <= XF)                                sp4f(at, t, r, half, x + (size_t)nidS * XF + c0);
      else if (lo >= XF && hi <= XF + 12)          sp4s(at, t, r, half, nsum + (size_t)nidS * 12 + (c0 - XF), rcnt);
      else if (lo >= XF + 12 && hi <= XF + 24)     sp4f(at, t, r, half, nmax + (size_t)nidS * 12 + (c0 - XF - 12));
      else if (lo >= XF + 24 && hi <= fin)         sp4f(at, t, r, half, uu + (size_t)g * UF + (c0 - XF - 24));
      else if (lo >= fin)                          sp0(at, t, r, half);
      else                                         spf(at, t, r, half, c0, nin);
    }
    asm volatile("s_wait_dscnt 0" ::: "memory");
    v8f d = mlp_tile<KC, true>(at, sH[wv], sW1, sB1, sW2, sB2, lane);
    if (nc < 12) {
#pragma unroll
      for (int v = 0; v < 8; ++v) {
        int m = v + 8 * half;
        int nn = base + m;
        if (nn < N) {
          float val = d[v];
          x_out[(size_t)nn * 12 + nc] = val;
          int gg = batch[nn];
          atomicAdd(&gsum[gg * 12 + nc], val);
          atomicMax((int*)&gmax[gg * 12 + nc], __float_as_int(val));
        }
      }
    }
    if (half == 0 && vn) atomicAdd(&gcnt[g], 1.0f);
  }
}

// Per-graph MLP on [u | mean | max]; GlobalModel and the final FC.
template <int UF, int KG, int FOUT, bool RELU2, bool SIG>
__global__ __launch_bounds__(TB) void graph_mlp_kernel(
    const float* __restrict__ uu, const float* __restrict__ gsum,
    const float* __restrict__ gmax, const float* __restrict__ gcnt,
    const float* __restrict__ W1, const float* __restrict__ B1,
    const float* __restrict__ W2, const float* __restrict__ B2,
    float* __restrict__ out, int G) {
  constexpr int KC = KG / 32;
  __shared__ __align__(32) _Float16 sW1[2 * KC * 512], sW2[512];
  __shared__ float sB1[32], sB2[16];
  __shared__ __align__(32) _Float16 sT[WPB][KC * 512];
  __shared__ __align__(32) _Float16 sH[WPB][512];
  stage_w1_frag(W1, UF + 24, KC, sW1);
  stage_w2_frag(W2, B1, B2, FOUT, sW2, sB1, sB2);
  __syncthreads();

  const int lane = threadIdx.x & 31, wv = threadIdx.x >> 5;
  const int r = lane & 15, half = lane >> 4, nc = lane & 15;
  const int ntiles = (G + 15) >> 4;
  _Float16* at = sT[wv];

  for (int tile = blockIdx.x * WPB + wv; tile < ntiles; tile += gridDim.x * WPB) {
    const int base = tile << 4;
    const int gid = base + r;
    const bool vg = (gid < G);
    const int gidS = vg ? gid : 0;
    float rgc = 1.0f / fmaxf(gcnt[gidS], 1.f);
    auto gin = [&](int c) -> float {
      if (c >= UF + 24) return 0.f;
      if (c < UF)       return uu[gidS * UF + c];
      if (c < UF + 12)  return gsum[gidS * 12 + (c - UF)] * rgc;
      return gmax[gidS * 12 + (c - UF - 12)];
    };
#pragma unroll
    for (int t = 0; t < KG / 8; ++t) {
      const int lo = 8 * t, hi = lo + 8;
      const int c0 = lo + 4 * half;
      if (hi <= UF)                           sp4f(at, t, r, half, uu + (size_t)gidS * UF + c0);
      else if (lo >= UF && hi <= UF + 12)     sp4s(at, t, r, half, gsum + (size_t)gidS * 12 + (c0 - UF), rgc);
      else if (lo >= UF + 12 && hi <= UF + 24) sp4f(at, t, r, half, gmax + (size_t)gidS * 12 + (c0 - UF - 12));
      else if (lo >= UF + 24)                 sp0(at, t, r, half);
      else                                    spf(at, t, r, half, c0, gin);
    }
    asm volatile("s_wait_dscnt 0" ::: "memory");
    v8f d = mlp_tile<KC, RELU2>(at, sH[wv], sW1, sB1, sW2, sB2, lane);
    if (nc < FOUT) {
#pragma unroll
      for (int v = 0; v < 8; ++v) {
        int m = v + 8 * half;
        int gg = base + m;
        if (gg < G) {
          float val = d[v];
          if (SIG) val = 1.f / (1.f + expf(-val));
          out[(size_t)gg * FOUT + nc] = val;
        }
      }
    }
  }
}

extern "C" void kernel_launch(void* const* d_in, const int* in_sizes, int n_in,
                              void* d_out, int out_size, void* d_ws, size_t ws_size,
                              hipStream_t stream) {
  (void)n_in; (void)out_size; (void)ws_size;
  const float* x0    = (const float*)d_in[0];
  const int*   ei    = (const int*)d_in[1];
  const float* ea0   = (const float*)d_in[2];
  const float* u0    = (const float*)d_in[3];
  const float* prm   = (const float*)d_in[4];
  const int*   batch = (const int*)d_in[5];
  // params pytree leaves, keys sorted: e1,e2,e3,fc,g1,g2,g3,n1a,n1b,n2a,n2b,n3a,n3b
  const float* P[52];
  for (int i = 0; i < 52; ++i) P[i] = (const float*)d_in[6 + i];
  const float* const* e1  = P + 0;  const float* const* e2  = P + 4;
  const float* const* e3  = P + 8;  const float* const* fc  = P + 12;
  const float* const* g1  = P + 16; const float* const* g2  = P + 20;
  const float* const* g3  = P + 24; const float* const* n1a = P + 28;
  const float* const* n1b = P + 32; const float* const* n2a = P + 36;
  const float* const* n2b = P + 40; const float* const* n3a = P + 44;
  const float* const* n3b = P + 48;

  const int N = in_sizes[0] / 8;
  const int E = in_sizes[1] / 2;
  const int G = in_sizes[3] / 3;

  float* ws = (float*)d_ws;
  float* u4  = ws; ws += (size_t)G * 4;
  float* u1  = ws; ws += (size_t)G * 9;
  float* u2  = ws; ws += (size_t)G * 9;
  float* u3  = ws; ws += (size_t)G * 9;
  float* ea1 = ws; ws += (size_t)E * 6;
  float* ea2 = ws; ws += (size_t)E * 6;   // ea3 reuses ea1
  float* x1  = ws; ws += (size_t)N * 12;
  float* x2  = ws; ws += (size_t)N * 12;  // x3 reuses x1
  float* nsum = ws; ws += (size_t)N * 12; // nsum|nmax|ncnt contiguous (one memset)
  float* nmax = ws; ws += (size_t)N * 12;
  float* ncnt = ws; ws += (size_t)N;
  float* gsum = ws; ws += (size_t)G * 12; // gsum|gmax|gcnt contiguous
  float* gmax = ws; ws += (size_t)G * 12;
  float* gcnt = ws; ws += (size_t)G;

  const int eBlocks = (((E + 15) >> 4) + WPB - 1) / WPB;
  const int nBlocks = (((N + 15) >> 4) + WPB - 1) / WPB;
  const int gBlocks = (((G + 15) >> 4) + WPB - 1) / WPB;

  build_u4_kernel<<<(G + 255) / 256, 256, 0, stream>>>(u0, prm, u4, G);

  // ---- Layer 1: node=8, edge=4, global=4 ----
  hipMemsetAsync(nsum, 0, (size_t)N * 25 * sizeof(float), stream);
  edge_layer_kernel<8, 4, 4, 32><<<eBlocks, TB, 0, stream>>>(
      x0, ei, ea0, u4, batch, e1[0], e1[1], e1[2], e1[3],
      n1a[0], n1a[1], n1a[2], n1a[3], ea1, nsum, nmax, ncnt, E);
  hipMemsetAsync(gsum, 0, (size_t)G * 25 * sizeof(float), stream);
  node_layer_kernel<8, 4, 64><<<nBlocks, TB, 0, stream>>>(
      x0, nsum, nmax, ncnt, u4, batch, n1b[0], n1b[1], n1b[2], n1b[3],
      x1, gsum, gmax, gcnt, N);
  graph_mlp_kernel<4, 32, 9, true, false><<<gBlocks, TB, 0, stream>>>(
      u4, gsum, gmax, gcnt, g1[0], g1[1], g1[2], g1[3], u1, G);

  // ---- Layer 2: node=12, edge=6, global=9 ----
  hipMemsetAsync(nsum, 0, (size_t)N * 25 * sizeof(float), stream);
  edge_layer_kernel<12, 6, 9, 64><<<eBlocks, TB, 0, stream>>>(
      x1, ei, ea1, u1, batch, e2[0], e2[1], e2[2], e2[3],
      n2a[0], n2a[1], n2a[2], n2a[3], ea2, nsum, nmax, ncnt, E);
  hipMemsetAsync(gsum, 0, (size_t)G * 25 * sizeof(float), stream);
  node_layer_kernel<12, 9, 64><<<nBlocks, TB, 0, stream>>>(
      x1, nsum, nmax, ncnt, u1, batch, n2b[0], n2b[1], n2b[2], n2b[3],
      x2, gsum, gmax, gcnt, N);
  graph_mlp_kernel<9, 64, 9, true, false><<<gBlocks, TB, 0, stream>>>(
      u1, gsum, gmax, gcnt, g2[0], g2[1], g2[2], g2[3], u2, G);

  // ---- Layer 3 ----
  hipMemsetAsync(nsum, 0, (size_t)N * 25 * sizeof(float), stream);
  edge_layer_kernel<12, 6, 9, 64><<<eBlocks, TB, 0, stream>>>(
      x2, ei, ea2, u2, batch, e3[0], e3[1], e3[2], e3[3],
      n3a[0], n3a[1], n3a[2], n3a[3], ea1, nsum, nmax, ncnt, E);
  hipMemsetAsync(gsum, 0, (size_t)G * 25 * sizeof(float), stream);
  node_layer_kernel<12, 9, 64><<<nBlocks, TB, 0, stream>>>(
      x2, nsum, nmax, ncnt, u2, batch, n3b[0], n3b[1], n3b[2], n3b[3],
      x1, gsum, gmax, gcnt, N);
  graph_mlp_kernel<9, 64, 9, true, false><<<gBlocks, TB, 0, stream>>>(
      u2, gsum, gmax, gcnt, g3[0], g3[1], g3[2], g3[3], u3, G);

  // ---- Final FC + sigmoid (reuses layer-3 graph aggregates of x3) ----
  graph_mlp_kernel<9, 64, 1, false, true><<<gBlocks, TB, 0, stream>>>(
      u3, gsum, gmax, gcnt, fc[0], fc[1], fc[2], fc[3], (float*)d_out, G);
}
